// CasualMultiheadDiffAttn_53884659696373
// MI455X (gfx1250) — matrix operand order repaired and checked by reading, not compiled, not run on hardware
//
#include <hip/hip_runtime.h>

// ---------------- problem constants ----------------
#define B_    2
#define L_    2048
#define E_    1024
#define NH_   16
#define HD_   64
#define HALF_ 32

static constexpr float kScale  = 0.17677669529663687f; // 1/sqrt(32)
static constexpr float kNegInf = -1e30f;

typedef __attribute__((ext_vector_type(16))) __bf16 v16bf;
typedef __attribute__((ext_vector_type(8)))  float  v8f;
typedef __attribute__((ext_vector_type(4)))  unsigned tdm_u4;
typedef __attribute__((ext_vector_type(8)))  int      tdm_i8;
typedef __attribute__((ext_vector_type(4)))  int      tdm_i4;

// ---------------- small helpers ----------------
static __device__ __forceinline__ __bf16 f2bf(float f) {
  union { float f; unsigned u; } v; v.f = f;
  unsigned r = v.u + 0x7FFFu + ((v.u >> 16) & 1u);          // round-to-nearest-even
  union { unsigned short s; __bf16 b; } o; o.s = (unsigned short)(r >> 16);
  return o.b;
}

static __device__ __forceinline__ v8f v8f_zero() {
  v8f z;
#pragma unroll
  for (int i = 0; i < 8; ++i) z[i] = 0.0f;
  return z;
}

static __device__ __forceinline__ v8f wmma_bf16(v16bf a, v16bf b, v8f c) {
  // D = A(16x32) * B(32x16) + C(16x16), fp32 accumulate
  return __builtin_amdgcn_wmma_f32_16x16x32_bf16(false, a, false, b, (short)0, c, false, false);
}

// LDS byte offset of a __shared__ pointer (flat shared addr: aperture in [63:32], offset in [31:0])
static __device__ __forceinline__ unsigned lds_off(const void* p) {
  return (unsigned)(uintptr_t)p;
}

// TDM: 2-D tile DMA load Global -> LDS.  rows x row_dwords dwords, contiguous rows.
// D# per CDNA5 ISA ch.8: group0 {count, lds_addr, global_addr, type=2},
// group1 {data_size=4B, tensor_dim0/1, tile_dim0/1, dim0_stride}; groups 2/3 zero (2-D).
static __device__ __forceinline__ void tdm_load_2d(unsigned lds_addr, const void* gaddr,
                                                   unsigned row_dwords, unsigned rows) {
  unsigned long long ga = (unsigned long long)(uintptr_t)gaddr;
  tdm_u4 g0;
  g0[0] = 1u;                                              // count=1, user mode, no gather
  g0[1] = lds_addr;                                        // lds_addr [63:32]
  g0[2] = (unsigned)(ga & 0xFFFFFFFFu);                    // global_addr [95:64]
  g0[3] = (unsigned)((ga >> 32) & 0x1FFFFFFu) | (2u << 30); // addr[56:32] | type=2 ("image")
  tdm_i8 g1;
  g1[0] = (int)(2u << 16);                                 // workgroup_mask=0, data_size=4B
  g1[1] = (int)((row_dwords & 0xFFFFu) << 16);             // tensor_dim0 [79:48] lo
  g1[2] = (int)((row_dwords >> 16) | ((rows & 0xFFFFu) << 16)); // dim0 hi | tensor_dim1 lo
  g1[3] = (int)((rows >> 16) | ((row_dwords & 0xFFFFu) << 16)); // dim1 hi | tile_dim0
  g1[4] = (int)(rows & 0xFFFFu);                           // tile_dim1 | tile_dim2=0
  g1[5] = (int)row_dwords;                                 // tensor_dim0_stride lo32
  g1[6] = 0;                                               // stride hi | dim1_stride lo (unused, 2-D)
  g1[7] = 0;
  tdm_i4 gz4 = {0, 0, 0, 0};
  tdm_i8 gz8 = {0, 0, 0, 0, 0, 0, 0, 0};
  // 6-arg toolchain variant: (g0, g1, g2, g3, extra, cpol)
  __builtin_amdgcn_tensor_load_to_lds(g0, g1, gz4, gz4, gz8, 0);
}

// A-operand fragment: 16(M) x 32(K) row-major tile, leading dim ld (bf16 elems).
// lanes 0-15 -> M=lane, K {0..15}; lanes 16-31 -> K {16..31} (matches frag_B).
static __device__ __forceinline__ v16bf frag_A(const __bf16* p, int ld, int lane) {
  int m  = lane & 15;
  int kb = (lane & 16) ? 16 : 0;
  const unsigned* q = (const unsigned*)(p + m * ld + kb);
  v16bf a;
  union { unsigned u; __bf16 b[2]; } t;
#pragma unroll
  for (int i = 0; i < 4; ++i) { t.u = q[i];     a[2*i]   = t.b[0]; a[2*i+1] = t.b[1]; }
#pragma unroll
  for (int i = 0; i < 4; ++i) { t.u = q[4 + i]; a[8+2*i] = t.b[0]; a[9+2*i] = t.b[1]; }
  return a;
}

// B-operand (32x16 = KxN), memory "N-major": B[k][n] = p[n*ld + k]  (NT layout)
static __device__ __forceinline__ v16bf frag_B_nt(const __bf16* p, int ld, int lane) {
  int n  = lane & 15;
  int kb = (lane & 16) ? 16 : 0;
  const unsigned* q = (const unsigned*)(p + n * ld + kb);
  v16bf b;
  union { unsigned u; __bf16 v[2]; } t;
#pragma unroll
  for (int i = 0; i < 8; ++i) { t.u = q[i]; b[2*i] = t.v[0]; b[2*i+1] = t.v[1]; }
  return b;
}

// B-operand (32x16 = KxN), memory K-major: B[k][n] = p[k*ld + n]
static __device__ __forceinline__ v16bf frag_B_nn(const __bf16* p, int ld, int lane) {
  int n  = lane & 15;
  int kb = (lane & 16) ? 16 : 0;
  v16bf b;
#pragma unroll
  for (int i = 0; i < 16; ++i) b[i] = p[(kb + i) * ld + n];
  return b;
}

// ---------------- kernel: fp32 -> bf16 ----------------
__global__ __launch_bounds__(256) void cvt_f32_bf16(const float* __restrict__ src,
                                                    __bf16* __restrict__ dst, int n) {
  int i = blockIdx.x * 256 + threadIdx.x;
  if (i < n) dst[i] = f2bf(src[i]);
}

// ---------------- kernel: lambda_full ----------------
__global__ void lambda_kernel(const float* __restrict__ lq1, const float* __restrict__ lk1,
                              const float* __restrict__ lq2, const float* __restrict__ lk2,
                              float* __restrict__ out) {
  int t = threadIdx.x;  // 32 threads
  float a = lq1[t] * lk1[t];
  float b = lq2[t] * lk2[t];
#pragma unroll
  for (int off = 1; off < 32; off <<= 1) {
    a += __shfl_xor(a, off, 32);
    b += __shfl_xor(b, off, 32);
  }
  if (t == 0) out[0] = expf(a) - expf(b) + 0.2f;   // LAMBDA_INIT = 0.2
}

// ---------------- kernel: tiled NT GEMM  Y[m,n] = sum_k A[m,k]*W[n,k] ----------------
// mode 0: bf16 -> Qh/Kh layout [b, 32, L, 32]
// mode 1: bf16 -> Vh layout [b, 16, L, 64]
// mode 2: fp32 flat [M, N]
__global__ __launch_bounds__(256) void gemm_nt(const __bf16* __restrict__ A,
                                               const __bf16* __restrict__ W,
                                               void* __restrict__ dst,
                                               int M, int N, int K, int mode) {
  __shared__ __align__(16) __bf16 As[128 * 32];
  __shared__ __align__(16) __bf16 Bs[64 * 32];
  const int tid  = threadIdx.x;
  const int lane = tid & 31;
  const int wave = tid >> 5;
  const int m0   = blockIdx.x * 128;
  const int n0   = blockIdx.y * 64;
  const int wm   = wave * 16;
  const int Kq   = K >> 3;                  // K in uint4 units
  const uint4* Ag4 = (const uint4*)A;
  const uint4* Wg4 = (const uint4*)W;
  uint4* As4 = (uint4*)As;
  uint4* Bs4 = (uint4*)Bs;

  v8f acc[4];
#pragma unroll
  for (int i = 0; i < 4; ++i) acc[i] = v8f_zero();

  const int nk = K >> 5;
  for (int kt = 0; kt < nk; ++kt) {
    const int kq = kt * 4;                  // k offset in uint4 units
    __syncthreads();
    // A tile: 128 rows x 4 uint4 (64B per row)
#pragma unroll
    for (int i = tid; i < 512; i += 256) {
      int r = i >> 2, c = i & 3;
      As4[i] = Ag4[(size_t)(m0 + r) * Kq + kq + c];
    }
    // B tile: 64 rows x 4 uint4
    {
      int r = tid >> 2, c = tid & 3;
      Bs4[tid] = Wg4[(size_t)(n0 + r) * Kq + kq + c];
    }
    // prefetch next A tile (cacheline-granular)
    if (kt + 1 < nk) {
      int r = tid >> 2, c = tid & 3;
      __builtin_prefetch(&Ag4[(size_t)(m0 + r) * Kq + kq + 4 + c], 0, 0);
    }
    __syncthreads();
    v16bf a = frag_A(As + wm * 32, 32, lane);
#pragma unroll
    for (int nt = 0; nt < 4; ++nt) {
      v16bf b = frag_B_nt(Bs + nt * 16 * 32, 32, lane);
      acc[nt] = wmma_bf16(a, b, acc[nt]);
    }
  }

  const int n_   = lane & 15;
  const int moff = (lane & 16) ? 8 : 0;
#pragma unroll
  for (int nt = 0; nt < 4; ++nt) {
    int gn = n0 + nt * 16 + n_;
#pragma unroll
    for (int r = 0; r < 8; ++r) {
      int gm = m0 + wm + moff + r;
      float v = acc[nt][r];
      if (mode == 0) {
        int bb = gm >> 11, ll = gm & 2047;
        int sh = gn >> 5,  d  = gn & 31;
        ((__bf16*)dst)[((size_t)(bb * 32 + sh) * L_ + ll) * HALF_ + d] = f2bf(v);
      } else if (mode == 1) {
        int bb = gm >> 11, ll = gm & 2047;
        int h  = gn >> 6,  d  = gn & 63;
        ((__bf16*)dst)[((size_t)(bb * NH_ + h) * L_ + ll) * HD_ + d] = f2bf(v);
      } else {
        ((float*)dst)[(size_t)gm * N + gn] = v;
      }
    }
  }
}

// ---------------- kernel: per-row softmax stats (pass 1, streaming) ----------------
// grid (L/64, 32 sub-heads, B); block 128 (4 waves, 16 q-rows each)
__global__ __launch_bounds__(128) void attn_stats(const __bf16* __restrict__ Qh,
                                                  const __bf16* __restrict__ Kh,
                                                  float* __restrict__ rowm,
                                                  float* __restrict__ rowl) {
  __shared__ __align__(16) __bf16 Qs[64 * 32];
  __shared__ __align__(16) __bf16 Ks[32 * 32];
  const int tid  = threadIdx.x;
  const int lane = tid & 31;
  const int wave = tid >> 5;
  const int qbase = blockIdx.x * 64;
  const size_t hr = (size_t)(blockIdx.z * 32 + blockIdx.y) * L_;
  const int wm   = wave * 16;
  const int n_   = lane & 15;
  const int moff = (lane & 16) ? 8 : 0;

  { const uint4* g = (const uint4*)(Qh + (hr + qbase) * HALF_);
    uint4* s = (uint4*)Qs;
#pragma unroll
    for (int i = tid; i < 256; i += 128) s[i] = g[i]; }
  __syncthreads();
  v16bf qa = frag_A(Qs + wm * 32, 32, lane);

  float mrun[8], lrun[8];
#pragma unroll
  for (int r = 0; r < 8; ++r) { mrun[r] = kNegInf; lrun[r] = 0.0f; }

  const int ktmax = (qbase + 63) >> 5;
  for (int kt = 0; kt <= ktmax; ++kt) {
    __syncthreads();
    { const uint4* g = (const uint4*)(Kh + (hr + kt * 32) * HALF_);
      uint4* s = (uint4*)Ks;
      if (tid < 128) s[tid] = g[tid]; }
    __syncthreads();
    v8f z  = v8f_zero();
    v8f s0 = wmma_bf16(qa, frag_B_nt(Ks,           32, lane), z);
    v8f s1 = wmma_bf16(qa, frag_B_nt(Ks + 16 * 32, 32, lane), z);
    const int k0g = kt * 32 + n_;
#pragma unroll
    for (int r = 0; r < 8; ++r) {
      int qg = qbase + wm + moff + r;
      float a0 = (k0g      <= qg) ? s0[r] * kScale : kNegInf;
      float a1 = (k0g + 16 <= qg) ? s1[r] * kScale : kNegInf;
      float t = fmaxf(a0, a1);
      t = fmaxf(t, __shfl_xor(t, 1, 32));
      t = fmaxf(t, __shfl_xor(t, 2, 32));
      t = fmaxf(t, __shfl_xor(t, 4, 32));
      t = fmaxf(t, __shfl_xor(t, 8, 32));
      float nm = fmaxf(mrun[r], t);
      float p  = __expf(a0 - nm) + __expf(a1 - nm);
      p += __shfl_xor(p, 1, 32);
      p += __shfl_xor(p, 2, 32);
      p += __shfl_xor(p, 4, 32);
      p += __shfl_xor(p, 8, 32);
      lrun[r] = lrun[r] * __expf(mrun[r] - nm) + p;
      mrun[r] = nm;
    }
  }
  if (n_ == 0) {   // lanes 0 (rows 0..7) and 16 (rows 8..15)
#pragma unroll
    for (int r = 0; r < 8; ++r) {
      int qg = qbase + wm + moff + r;
      rowm[hr + qg] = mrun[r];
      rowl[hr + qg] = lrun[r];
    }
  }
}

// ---------------- kernel: differential attention A + O (pass 2) ----------------
// grid (L/128, 16 head pairs, B); block 256 (8 waves, 16 q-rows each)
// K/V tiles streamed by the Tensor Data Mover, double-buffered (TENSORcnt).
__global__ __launch_bounds__(256) void attn_av(const __bf16* __restrict__ Qh,
                                               const __bf16* __restrict__ Kh,
                                               const __bf16* __restrict__ Vh,
                                               const float* __restrict__ rowm,
                                               const float* __restrict__ rowl,
                                               const float* __restrict__ lambda_p,
                                               float* __restrict__ Aout,
                                               __bf16* __restrict__ Oh) {
  __shared__ __align__(16) __bf16 Q0s[128 * 32];
  __shared__ __align__(16) __bf16 Q1s[128 * 32];
  __shared__ __align__(16) __bf16 K0s[2][32 * 32];   // double-buffered TDM targets
  __shared__ __align__(16) __bf16 K1s[2][32 * 32];
  __shared__ __align__(16) __bf16 Vs[2][32 * 64];
  __shared__ __align__(16) __bf16 Ast[8][16 * 32];   // bf16 A for P@V WMMA
  __shared__ __align__(16) float  Afp[8][16 * 32];   // fp32 A for coalesced output

  const int tid  = threadIdx.x;
  const int lane = tid & 31;
  const int wave = tid >> 5;
  const int qbase = blockIdx.x * 128;
  const int h  = blockIdx.y;
  const int bz = blockIdx.z;
  const size_t hr0 = (size_t)(bz * 32 + 2 * h)     * L_;
  const size_t hr1 = (size_t)(bz * 32 + 2 * h + 1) * L_;
  const size_t hv  = (size_t)(bz * NH_ + h)        * L_;
  const int wm   = wave * 16;
  const int n_   = lane & 15;
  const int moff = (lane & 16) ? 8 : 0;
  const float lam = *lambda_p;

  { const uint4* g0 = (const uint4*)(Qh + (hr0 + qbase) * HALF_);
    const uint4* g1 = (const uint4*)(Qh + (hr1 + qbase) * HALF_);
    uint4* s0 = (uint4*)Q0s; uint4* s1 = (uint4*)Q1s;
#pragma unroll
    for (int i = tid; i < 512; i += 256) { s0[i] = g0[i]; s1[i] = g1[i]; } }
  __syncthreads();
  v16bf qa0 = frag_A(Q0s + wm * 32, 32, lane);
  v16bf qa1 = frag_A(Q1s + wm * 32, 32, lane);

  float m0v[8], il0[8], m1v[8], il1[8];
#pragma unroll
  for (int r = 0; r < 8; ++r) {
    int qg = qbase + wm + moff + r;
    m0v[r] = rowm[hr0 + qg]; il0[r] = 1.0f / rowl[hr0 + qg];
    m1v[r] = rowm[hr1 + qg]; il1[r] = 1.0f / rowl[hr1 + qg];
  }

  v8f oacc[4];
#pragma unroll
  for (int i = 0; i < 4; ++i) oacc[i] = v8f_zero();

  __bf16* Aw = Ast[wave];
  float*  Af = Afp[wave];
  const int ktmax = (qbase + 127) >> 5;

  // TDM issue helper: one batch = K0,K1,V tiles for key-tile kt2 into buffer buf2
  auto tdm_batch = [&](int kt2, int buf2) {
    tdm_load_2d(lds_off(&K0s[buf2][0]), Kh + (hr0 + (size_t)kt2 * 32) * HALF_, 16u, 32u);
    tdm_load_2d(lds_off(&K1s[buf2][0]), Kh + (hr1 + (size_t)kt2 * 32) * HALF_, 16u, 32u);
    tdm_load_2d(lds_off(&Vs[buf2][0]),  Vh + (hv  + (size_t)kt2 * 32) * HD_,   32u, 32u);
  };

  if (wave == 0) tdm_batch(0, 0);          // prologue: start DMA for tile 0

  for (int kt = 0; kt <= ktmax; ++kt) {
    const int buf = kt & 1;
    __syncthreads();                        // all waves done reading buf^1 (from kt-1)
    if (wave == 0) {
      if (kt < ktmax) {
        tdm_batch(kt + 1, buf ^ 1);         // overlap DMA of next tile with this tile's math
        __builtin_amdgcn_s_wait_tensorcnt(3);  // in-order: batch kt complete
      } else {
        __builtin_amdgcn_s_wait_tensorcnt(0);
      }
    }
    __syncthreads();                        // tile kt visible to all waves

    const __bf16* k0t = K0s[buf];
    const __bf16* k1t = K1s[buf];
    const __bf16* vt  = Vs[buf];
    v8f z   = v8f_zero();
    v8f s00 = wmma_bf16(qa0, frag_B_nt(k0t,           32, lane), z);
    v8f s01 = wmma_bf16(qa0, frag_B_nt(k0t + 16 * 32, 32, lane), z);
    v8f s10 = wmma_bf16(qa1, frag_B_nt(k1t,           32, lane), z);
    v8f s11 = wmma_bf16(qa1, frag_B_nt(k1t + 16 * 32, 32, lane), z);
#pragma unroll
    for (int nt = 0; nt < 2; ++nt) {
      v8f sA = nt ? s01 : s00;
      v8f sB = nt ? s11 : s10;
      int kg = kt * 32 + nt * 16 + n_;
#pragma unroll
      for (int r = 0; r < 8; ++r) {
        int qg = qbase + wm + moff + r;
        float a = 0.0f;
        if (kg <= qg) {
          float p0 = __expf(sA[r] * kScale - m0v[r]) * il0[r];
          float p1 = __expf(sB[r] * kScale - m1v[r]) * il1[r];
          a = p0 - lam * p1;
        }
        Af[(moff + r) * 32 + nt * 16 + n_] = a;        // fp32 for output
        Aw[(moff + r) * 32 + nt * 16 + n_] = f2bf(a);  // bf16 for P@V
      }
    }
    // coalesced A output: 16 rows x 32 cols fp32 per wave -> 4 float4 per lane
    {
      const int rowbase = qbase + wm;
#pragma unroll
      for (int j = 0; j < 4; ++j) {
        int f  = lane + j * 32;        // 0..127 float4 slots
        int rr = f >> 3;               // 8 float4 per row
        int cc = (f & 7) << 2;
        float4 v = *(const float4*)(Af + rr * 32 + cc);
        *(float4*)(Aout + (hv + rowbase + rr) * L_ + kt * 32 + cc) = v;
      }
    }
    v16bf av = frag_A(Aw, 32, lane);
#pragma unroll
    for (int nt = 0; nt < 4; ++nt) {
      v16bf bv = frag_B_nn(vt + nt * 16, 64, lane);
      oacc[nt] = wmma_bf16(av, bv, oacc[nt]);
    }
  }

  // zero-fill A columns [qbase+128, L) for this 128-row tile (strict upper triangle)
  {
    int kstart = qbase + 128;
    int zq = L_ - kstart;
    if (zq > 0) {
      int zq4 = zq >> 2;
      float4 z4; z4.x = z4.y = z4.z = z4.w = 0.0f;
      for (int i = tid; i < 128 * zq4; i += 256) {
        int rr = i / zq4, cc = (i - rr * zq4) << 2;
        *(float4*)(Aout + (hv + qbase + rr) * L_ + kstart + cc) = z4;
      }
    }
  }

  // RMSNorm over head_dim (64), scale by (1 - 0.2), store bf16 O merged-heads
#pragma unroll
  for (int r = 0; r < 8; ++r) {
    float ss = 0.0f;
#pragma unroll
    for (int nt = 0; nt < 4; ++nt) ss += oacc[nt][r] * oacc[nt][r];
    ss += __shfl_xor(ss, 1, 32);
    ss += __shfl_xor(ss, 2, 32);
    ss += __shfl_xor(ss, 4, 32);
    ss += __shfl_xor(ss, 8, 32);
    float g = rsqrtf(ss * (1.0f / 64.0f) + 1e-5f) * 0.8f;
    int qg = qbase + wm + moff + r;
    size_t base = ((size_t)bz * L_ + qg) * E_ + h * HD_;
#pragma unroll
    for (int nt = 0; nt < 4; ++nt)
      Oh[base + nt * 16 + n_] = f2bf(oacc[nt][r] * g);
  }
}

// ---------------- host orchestration ----------------
extern "C" void kernel_launch(void* const* d_in, const int* in_sizes, int n_in,
                              void* d_out, int out_size, void* d_ws, size_t ws_size,
                              hipStream_t stream) {
  (void)in_sizes; (void)n_in; (void)out_size; (void)ws_size;
  const float* x   = (const float*)d_in[0];
  const float* Wq  = (const float*)d_in[1];
  const float* Wk  = (const float*)d_in[2];
  const float* Wv  = (const float*)d_in[3];
  const float* Wo  = (const float*)d_in[4];
  const float* lq1 = (const float*)d_in[5];
  const float* lk1 = (const float*)d_in[6];
  const float* lq2 = (const float*)d_in[7];
  const float* lk2 = (const float*)d_in[8];

  char* ws = (char*)d_ws;
  size_t off = 0;
  auto alloc = [&](size_t bytes) -> char* {
    char* p = ws + off;
    off += (bytes + 255) & ~(size_t)255;
    return p;
  };

  const size_t nX = (size_t)B_ * L_ * E_;   // 4,194,304
  const size_t nW = (size_t)E_ * E_;        // 1,048,576
  float*  lam  = (float*)alloc(256);
  __bf16* xb   = (__bf16*)alloc(nX * 2);
  __bf16* Wqb  = (__bf16*)alloc(nW * 2);
  __bf16* Wkb  = (__bf16*)alloc(nW * 2);
  __bf16* Wvb  = (__bf16*)alloc(nW * 2);
  __bf16* Wob  = (__bf16*)alloc(nW * 2);
  __bf16* Qh   = (__bf16*)alloc(nX * 2);
  __bf16* Kh   = (__bf16*)alloc(nX * 2);
  __bf16* Vh   = (__bf16*)alloc(nX * 2);
  __bf16* Oh   = (__bf16*)alloc(nX * 2);
  float* rowm  = (float*)alloc((size_t)B_ * 2 * NH_ * L_ * 4);
  float* rowl  = (float*)alloc((size_t)B_ * 2 * NH_ * L_ * 4);

  cvt_f32_bf16<<<dim3((unsigned)((nX + 255) / 256)), 256, 0, stream>>>(x,  xb,  (int)nX);
  cvt_f32_bf16<<<dim3((unsigned)((nW + 255) / 256)), 256, 0, stream>>>(Wq, Wqb, (int)nW);
  cvt_f32_bf16<<<dim3((unsigned)((nW + 255) / 256)), 256, 0, stream>>>(Wk, Wkb, (int)nW);
  cvt_f32_bf16<<<dim3((unsigned)((nW + 255) / 256)), 256, 0, stream>>>(Wv, Wvb, (int)nW);
  cvt_f32_bf16<<<dim3((unsigned)((nW + 255) / 256)), 256, 0, stream>>>(Wo, Wob, (int)nW);
  lambda_kernel<<<1, 32, 0, stream>>>(lq1, lk1, lq2, lk2, lam);

  dim3 gg(4096 / 128, 1024 / 64);
  gemm_nt<<<gg, 256, 0, stream>>>(xb, Wqb, (void*)Qh, 4096, 1024, 1024, 0);
  gemm_nt<<<gg, 256, 0, stream>>>(xb, Wkb, (void*)Kh, 4096, 1024, 1024, 0);
  gemm_nt<<<gg, 256, 0, stream>>>(xb, Wvb, (void*)Vh, 4096, 1024, 1024, 1);

  attn_stats<<<dim3(L_ / 64, 2 * NH_, B_), 128, 0, stream>>>(Qh, Kh, rowm, rowl);

  float* Aout = (float*)d_out + nX;   // A region follows O region
  attn_av<<<dim3(L_ / 128, NH_, B_), 256, 0, stream>>>(Qh, Kh, Vh, rowm, rowl, lam, Aout, Oh);

  gemm_nt<<<gg, 256, 0, stream>>>(Oh, Wob, d_out, 4096, 1024, 1024, 2);
}
